// HeteroGIN_2276332667317
// MI455X (gfx1250) — matrix operand order repaired and checked by reading, not compiled, not run on hardware
//
#include <hip/hip_runtime.h>
#include <math.h>

// ---------------- problem constants ----------------
constexpr int kNG   = 4096;      // graphs
constexpr int kP    = 32;        // nodes per graph (fully connected)
constexpr int kN    = kNG * kP;  // 131072 nodes
constexpr int kDin  = 7;
constexpr int kDkin = 8;
constexpr int kH    = 32;
constexpr int kH2   = 128;
constexpr float kC0 = 30.0f / 31.0f;   // h' = C0*h + C1*graphsum  (mean agg, deg=31)
constexpr float kC1 = 1.0f  / 31.0f;

typedef __attribute__((ext_vector_type(16))) _Float16 v16h;
typedef __attribute__((ext_vector_type(8)))  float    v8f;

// ---- WMMA operand maps (CDNA5 ISA 7.12.2, wave32) ----
// 16-bit A 16x32: lanes 0-15 row M=lane,   K in {0..7,16..23}  (kb=0)
//                 lanes 16-31 row M=lane-16,K in {8..15,24..31} (kb=8)
// Per lane that is two contiguous 8-float runs -> four aligned float4 loads.
// 16-bit B 32x16: lanes 0-15 col=lane K in {0..7,16..23}; lanes 16-31 col=lane-16 K in {8..15,24..31}.
// B is pre-packed into this layout (f16) so GEMMs load 32 contiguous bytes/lane.

// ---------------- one-off weight pre-pack into WMMA-B layout ----------------
// grid (ctiles, kchunks), block 32. dst tile index = ct*kchunks + kk.
__global__ void packW_k(const float* __restrict__ W, int dinK, int dout,
                        _Float16* __restrict__ dst) {
  const int ct = blockIdx.x, kk = blockIdx.y, lane = threadIdx.x;
  const int col = ct * 16 + (lane & 15);
  const int kb  = kk * 32 + ((lane < 16) ? 0 : 8);
  const int cc  = (col < dout) ? col : (dout - 1);
  v16h b;
#pragma unroll
  for (int i = 0; i < 16; ++i) {
    int k  = kb + ((i < 8) ? i : i + 8);
    int kc = (k < dinK) ? k : (dinK - 1);
    float v = W[(size_t)kc * dout + cc];          // clamped (always in-bounds) load
    b[i] = (_Float16)((k < dinK && col < dout) ? v : 0.f);  // select, no branch
  }
  ((v16h*)dst)[(size_t)(blockIdx.x * gridDim.y + blockIdx.y) * 32 + lane] = b;
}

// ---------------- fused GEMM (dout=32, K=32) + BN-stats epilogue ----------------
// MODE 0: A = C0*src + C1*gsum[graph]  (GIN mean-agg fused; src is dense [N,32])
// MODE 1: A = relu(src*scale + shift)  (previous BN+ReLU fused)
// MODE 2: A = src                      (pre-padded layer-0 input)
template <int MODE>
__global__ __launch_bounds__(256) void gemm32_k(
    const float* __restrict__ src, const float* __restrict__ gsum,
    const float* __restrict__ scale, const float* __restrict__ shift,
    const _Float16* __restrict__ BPk, const float* __restrict__ bias,
    float* __restrict__ out, float* __restrict__ partials) {
  const int wave = threadIdx.x >> 5;
  const int lane = threadIdx.x & 31;
  const int lh   = lane & 15;
  const int rt   = blockIdx.x * 8 + wave;
  const int row0 = rt * 16;
  const int row  = row0 + lh;
  const int q    = (lane < 16) ? 0 : 2;   // kb/4

  // ---- A operand: 4 unconditional float4 loads, transform in registers ----
  const float4* hr4 = (const float4*)(src + (size_t)row * kH);
  float4 h0 = hr4[q], h1 = hr4[q + 1], h2 = hr4[q + 4], h3 = hr4[q + 5];
  float f[16] = {h0.x, h0.y, h0.z, h0.w, h1.x, h1.y, h1.z, h1.w,
                 h2.x, h2.y, h2.z, h2.w, h3.x, h3.y, h3.z, h3.w};
  if (MODE == 0) {
    const float4* gr4 = (const float4*)(gsum + (size_t)(row >> 5) * kH);
    float4 g0 = gr4[q], g1 = gr4[q + 1], g2 = gr4[q + 4], g3 = gr4[q + 5];
    float g[16] = {g0.x, g0.y, g0.z, g0.w, g1.x, g1.y, g1.z, g1.w,
                   g2.x, g2.y, g2.z, g2.w, g3.x, g3.y, g3.z, g3.w};
#pragma unroll
    for (int i = 0; i < 16; ++i) f[i] = kC0 * f[i] + kC1 * g[i];
  } else if (MODE == 1) {
    const float4* sc4 = (const float4*)scale;
    const float4* sh4 = (const float4*)shift;
    float4 s0 = sc4[q], s1 = sc4[q + 1], s2 = sc4[q + 4], s3 = sc4[q + 5];
    float4 t0 = sh4[q], t1 = sh4[q + 1], t2 = sh4[q + 4], t3 = sh4[q + 5];
    float s[16] = {s0.x, s0.y, s0.z, s0.w, s1.x, s1.y, s1.z, s1.w,
                   s2.x, s2.y, s2.z, s2.w, s3.x, s3.y, s3.z, s3.w};
    float t[16] = {t0.x, t0.y, t0.z, t0.w, t1.x, t1.y, t1.z, t1.w,
                   t2.x, t2.y, t2.z, t2.w, t3.x, t3.y, t3.z, t3.w};
#pragma unroll
    for (int i = 0; i < 16; ++i) f[i] = fmaxf(f[i] * s[i] + t[i], 0.f);
  }
  v16h a;
#pragma unroll
  for (int i = 0; i < 16; ++i) a[i] = (_Float16)f[i];

  // ---- B operand: pre-packed, contiguous 32B per lane per tile ----
  const v16h* BP = (const v16h*)BPk;
  v16h b0 = BP[lane];
  v16h b1 = BP[32 + lane];

  v8f c0 = {}, c1 = {};
  c0 = __builtin_amdgcn_wmma_f32_16x16x32_f16(false, a, false, b0, (short)0, c0, false, false);
  c1 = __builtin_amdgcn_wmma_f32_16x16x32_f16(false, a, false, b1, (short)0, c1, false, false);

  const float bb0 = bias[lh], bb1 = bias[16 + lh];
  const int rb = (lane < 16) ? 0 : 8;   // C/D layout: lanes>=16 hold M=8..15
  float s0 = 0, q0 = 0, s1 = 0, q1 = 0;
#pragma unroll
  for (int r = 0; r < 8; ++r) {
    float v0 = c0[r] + bb0, v1 = c1[r] + bb1;
    out[(size_t)(row0 + rb + r) * kH + lh]      = v0;
    out[(size_t)(row0 + rb + r) * kH + 16 + lh] = v1;
    s0 += v0; q0 += v0 * v0; s1 += v1; q1 += v1 * v1;
  }
  // deterministic column stats: pair-combine halves, LDS, fixed-order wave sum
  s0 += __shfl_xor(s0, 16); q0 += __shfl_xor(q0, 16);
  s1 += __shfl_xor(s1, 16); q1 += __shfl_xor(q1, 16);
  __shared__ float part[8][64];
  if (lane < 16) {
    part[wave][lh] = s0;      part[wave][16 + lh] = s1;
    part[wave][32 + lh] = q0; part[wave][48 + lh] = q1;
  }
  __syncthreads();
  if (threadIdx.x < 64) {
    float t = 0.f;
#pragma unroll
    for (int w = 0; w < 8; ++w) t += part[w][threadIdx.x];
    partials[(size_t)blockIdx.x * 64 + threadIdx.x] = t;
  }
}

// ---------------- fused GEMM, K=128 (final MLP), optional stats ----------------
// MODE 0: A = src ; MODE 1: A = relu(src*scale+shift). All loads unconditional.
template <int MODE, bool STATS>
__global__ __launch_bounds__(256) void gemm128_k(
    const float* __restrict__ src,
    const float* __restrict__ scale, const float* __restrict__ shift,
    const _Float16* __restrict__ BPk, const float* __restrict__ bias,
    float* __restrict__ out, int out_stride, int dout, int dstore, int nct,
    float* __restrict__ partials) {
  __shared__ float part[8][256];
  const int wave = threadIdx.x >> 5;
  const int lane = threadIdx.x & 31;
  const int lh   = lane & 15;
  const int rt   = blockIdx.x * 8 + wave;
  const int row0 = rt * 16, row = row0 + lh;
  const int q    = (lane < 16) ? 0 : 2;
  const float4* hr4 = (const float4*)(src + (size_t)row * kH2);
  const float4* sc4 = (const float4*)scale;
  const float4* sh4 = (const float4*)shift;

  v16h a[4];
#pragma unroll
  for (int kk = 0; kk < 4; ++kk) {
    const int b4 = kk * 8 + q;
    float4 h0 = hr4[b4], h1 = hr4[b4 + 1], h2 = hr4[b4 + 4], h3 = hr4[b4 + 5];
    float f[16] = {h0.x, h0.y, h0.z, h0.w, h1.x, h1.y, h1.z, h1.w,
                   h2.x, h2.y, h2.z, h2.w, h3.x, h3.y, h3.z, h3.w};
    if (MODE == 1) {
      float4 s0 = sc4[b4], s1 = sc4[b4 + 1], s2 = sc4[b4 + 4], s3 = sc4[b4 + 5];
      float4 t0 = sh4[b4], t1 = sh4[b4 + 1], t2 = sh4[b4 + 4], t3 = sh4[b4 + 5];
      float s[16] = {s0.x, s0.y, s0.z, s0.w, s1.x, s1.y, s1.z, s1.w,
                     s2.x, s2.y, s2.z, s2.w, s3.x, s3.y, s3.z, s3.w};
      float t[16] = {t0.x, t0.y, t0.z, t0.w, t1.x, t1.y, t1.z, t1.w,
                     t2.x, t2.y, t2.z, t2.w, t3.x, t3.y, t3.z, t3.w};
#pragma unroll
      for (int i = 0; i < 16; ++i) f[i] = fmaxf(f[i] * s[i] + t[i], 0.f);
    }
#pragma unroll
    for (int i = 0; i < 16; ++i) a[kk][i] = (_Float16)f[i];
  }

  const v16h* BP = (const v16h*)BPk;
  const int rb = (lane < 16) ? 0 : 8;
  for (int ct = 0; ct < nct; ++ct) {
    v8f c = {};
#pragma unroll
    for (int kk = 0; kk < 4; ++kk) {
      v16h b = BP[(size_t)(ct * 4 + kk) * 32 + lane];
      c = __builtin_amdgcn_wmma_f32_16x16x32_f16(false, a[kk], false, b, (short)0, c, false, false);
    }
    int col = ct * 16 + lh;
    float bb = (col < dout) ? bias[col] : 0.f;
    float s = 0, qq = 0;
#pragma unroll
    for (int r = 0; r < 8; ++r) {
      float v = c[r] + bb;
      if (col < dstore) out[(size_t)(row0 + rb + r) * out_stride + col] = v;
      s += v; qq += v * v;
    }
    if (STATS) {
      s += __shfl_xor(s, 16); qq += __shfl_xor(qq, 16);
      if (lane < 16) { part[wave][col] = s; part[wave][128 + col] = qq; }
    }
  }
  if (STATS) {
    __syncthreads();
    float t = 0.f;
#pragma unroll
    for (int w = 0; w < 8; ++w) t += part[w][threadIdx.x];
    partials[(size_t)blockIdx.x * 256 + threadIdx.x] = t;
  }
}

// ---------------- prediction head: score = sum_l (gsum_l/32) @ W_l + b_l ----------------
// gsum_0 is zero-padded to 32 cols, weights zero-padded at pack time -> no guards.
__global__ __launch_bounds__(256) void pred_k(
    const float* __restrict__ gs,        // [5][NG*32]
    const _Float16* __restrict__ BPk,    // [5][8 tiles]
    const float* __restrict__ b0, const float* __restrict__ b1,
    const float* __restrict__ b2, const float* __restrict__ b3,
    const float* __restrict__ b4,
    float* __restrict__ hcat) {
  const float* Bs[5] = {b0, b1, b2, b3, b4};
  const int wave = threadIdx.x >> 5, lane = threadIdx.x & 31, lh = lane & 15;
  const int rt = blockIdx.x * 8 + wave;
  const int row0 = rt * 16, row = row0 + lh;   // row = graph id
  const int q = (lane < 16) ? 0 : 2;

  v16h a[5];
#pragma unroll
  for (int l = 0; l < 5; ++l) {
    const float4* gr4 = (const float4*)(gs + (size_t)l * kN + (size_t)row * kH);
    float4 h0 = gr4[q], h1 = gr4[q + 1], h2 = gr4[q + 4], h3 = gr4[q + 5];
    float f[16] = {h0.x, h0.y, h0.z, h0.w, h1.x, h1.y, h1.z, h1.w,
                   h2.x, h2.y, h2.z, h2.w, h3.x, h3.y, h3.z, h3.w};
#pragma unroll
    for (int i = 0; i < 16; ++i) a[l][i] = (_Float16)(f[i] * (1.f / 32.f));
  }
  const v16h* BP = (const v16h*)BPk;
  const int rb = (lane < 16) ? 0 : 8;
  for (int ct = 0; ct < 8; ++ct) {
    v8f c = {};
#pragma unroll
    for (int l = 0; l < 5; ++l) {
      v16h b = BP[(size_t)(l * 8 + ct) * 32 + lane];
      c = __builtin_amdgcn_wmma_f32_16x16x32_f16(false, a[l], false, b, (short)0, c, false, false);
    }
    int col = ct * 16 + lh;
    if (col < kH2 - kDkin) {
      float bb = 0.f;
#pragma unroll
      for (int l = 0; l < 5; ++l) bb += Bs[l][col];
#pragma unroll
      for (int r = 0; r < 8; ++r)
        hcat[(size_t)(row0 + rb + r) * kH2 + col] = c[r] + bb;
    }
  }
}

// ---------------- small deterministic helpers ----------------
__global__ void bn_reduce_k(const float* __restrict__ partials, int nparts, int D, float inv_n,
                            const float* __restrict__ gamma, const float* __restrict__ beta,
                            float* __restrict__ scale, float* __restrict__ shift) {
  __shared__ float acc[256];
  int t = threadIdx.x;  // 0..2D-1 (sums then sumsq)
  float s = 0.f;
  for (int w = 0; w < nparts; ++w) s += partials[(size_t)w * 2 * D + t];
  acc[t] = s;
  __syncthreads();
  if (t < D) {
    float mu  = acc[t] * inv_n;
    float var = acc[D + t] * inv_n - mu * mu;   // biased variance (training BN)
    float sc  = gamma[t] * rsqrtf(var + 1e-5f);
    scale[t] = sc;
    shift[t] = beta[t] - mu * sc;
  }
}

// stats of t = relu(z*s+sh) without materializing t (needed by the outer BN)
__global__ __launch_bounds__(256) void relu_stats_k(
    const float* __restrict__ z, const float* __restrict__ scale,
    const float* __restrict__ shift, float* __restrict__ partials, int rows_per_block) {
  int c = threadIdx.x & 31, rl = threadIdx.x >> 5;
  int r0 = blockIdx.x * rows_per_block;
  float sc = scale[c], sh = shift[c];
  float s = 0, q = 0;
  for (int r = r0 + rl; r < r0 + rows_per_block; r += 8) {
    float v = fmaxf(z[(size_t)r * kH + c] * sc + sh, 0.f);
    s += v; q += v * v;
  }
  __shared__ float ss[256], qq[256];
  ss[threadIdx.x] = s; qq[threadIdx.x] = q;
  __syncthreads();
  if (threadIdx.x < 64) {
    int cc = threadIdx.x & 31;
    float t = 0.f;
    if (threadIdx.x < 32) { for (int w = 0; w < 8; ++w) t += ss[w * 32 + cc]; }
    else                  { for (int w = 0; w < 8; ++w) t += qq[w * 32 + cc]; }
    partials[(size_t)blockIdx.x * 64 + threadIdx.x] = t;
  }
}

// h_next = relu( relu(z*s3+sh3)*s4 + sh4 )
__global__ void finalize_k(const float* __restrict__ z,
                           const float* __restrict__ s3, const float* __restrict__ h3,
                           const float* __restrict__ s4, const float* __restrict__ h4,
                           float* __restrict__ hout) {
  size_t i = (size_t)blockIdx.x * 256 + threadIdx.x;
  int c = (int)(i & 31);
  float v = fmaxf(z[i] * s3[c] + h3[c], 0.f);
  v = fmaxf(v * s4[c] + h4[c], 0.f);
  hout[i] = v;
}

// per-graph sums, zero-padded to 32 cols (mean agg + pred avg-pooling)
__global__ void graphsum_k(const float* __restrict__ h, int stride, int din,
                           float* __restrict__ gs) {
  int idx = blockIdx.x * 256 + threadIdx.x;  // < NG*32
  int g = idx >> 5, c = idx & 31;
  float s = 0.f;
  if (c < din) {
    const float* p = h + (size_t)g * kP * stride + c;
    for (int n = 0; n < kP; ++n) s += p[(size_t)n * stride];
  }
  gs[idx] = s;
}

// layer-0 prep: XA = C0*x + C1*gsum0, zero-padded [N,7] -> [N,32]
__global__ void xprep_k(const float* __restrict__ x, const float* __restrict__ gs0,
                        float* __restrict__ xa) {
  size_t i = (size_t)blockIdx.x * 256 + threadIdx.x;  // < N*32
  int c = (int)(i & 31);
  int row = (int)(i >> 5);
  float v = 0.f;
  if (c < kDin) v = kC0 * x[(size_t)row * kDin + c] + kC1 * gs0[(size_t)(row >> 5) * kH + c];
  xa[i] = v;
}

// per-graph max of kinematics -> hcat[:,120:128]
__global__ void kinmax_k(const float* __restrict__ kin, float* __restrict__ hcat) {
  int idx = blockIdx.x * 256 + threadIdx.x;  // < NG*8
  int g = idx >> 3, f = idx & 7;
  float m = kin[(size_t)g * kP * kDkin + f];
  for (int n = 1; n < kP; ++n) m = fmaxf(m, kin[(size_t)(g * kP + n) * kDkin + f]);
  hcat[(size_t)g * kH2 + (kH2 - kDkin) + f] = m;
}

// ---------------- host orchestration ----------------
extern "C" void kernel_launch(void* const* d_in, const int* in_sizes, int n_in,
                              void* d_out, int out_size, void* d_ws, size_t ws_size,
                              hipStream_t stream) {
  (void)in_sizes; (void)n_in; (void)out_size; (void)ws_size;
  const float* x   = (const float*)d_in[0];   // [N,7]
  const float* kin = (const float*)d_in[1];   // [N,8]
  // params flattened depth-first in setup_inputs() order, starting at d_in[2]:
  //  gin[l] l=0..3: w1,b1,w2,b2,w3,b3, bn1g,bn1b, bn2g,bn2b, apg,apb, outg,outb (14 each)
  //  pred[l] l=0..4: w,b (10) ; final: fw1,fb1,fw2,fb2,fw3,fb3, fbn1g..fbn2b (10)
  auto Pm = [&](int i) { return (const float*)d_in[2 + i]; };

  float* ws   = (float*)d_ws;
  float* Hbuf = ws;                       // [N,32]
  float* ZA   = ws + (size_t)kN * kH;     // [N,32]
  float* ZB   = ZA + (size_t)kN * kH;     // [N,32]  (also holds layer-0 padded input)
  float* GS   = ZB + (size_t)kN * kH;     // [5][NG,32]
  float* HCAT = GS + (size_t)5 * kN;      // [NG,128]
  float* ZF1  = HCAT + (size_t)kNG * kH2; // [NG,128]
  float* ZF2  = ZF1 + (size_t)kNG * kH2;  // [NG,128]
  float* PART = ZF2 + (size_t)kNG * kH2;  // 1024*64 floats
  float* SCA  = PART + 1024 * 64;
  float* SHA  = SCA + kH2;
  float* SCB  = SHA + kH2;
  float* SHB  = SCB + kH2;
  _Float16* WPK = (_Float16*)(SHB + kH2); // 132 tiles * 512 halves = 132 KB

  // packed-weight tile bases (tile = 512 halves = 32 lanes * v16h)
  auto tileP = [&](int t) { return WPK + (size_t)t * 512; };
  auto ginB  = [&](int l, int m) { return (l * 3 + m) * 2; };  // 2 tiles per 32x32 mat
  const int predB = 24, f1B = 64, f2B = 96, f3B = 128;

  // ---- pre-pack all weights into WMMA-B layout (one-off, tiny) ----
  for (int l = 0; l < 4; ++l) {
    int din0 = (l == 0) ? kDin : kH;
    packW_k<<<dim3(2, 1), 32, 0, stream>>>(Pm(l * 14 + 0), din0, kH, tileP(ginB(l, 0)));
    packW_k<<<dim3(2, 1), 32, 0, stream>>>(Pm(l * 14 + 2), kH,   kH, tileP(ginB(l, 1)));
    packW_k<<<dim3(2, 1), 32, 0, stream>>>(Pm(l * 14 + 4), kH,   kH, tileP(ginB(l, 2)));
  }
  for (int l = 0; l < 5; ++l) {
    int dinl = (l == 0) ? kDin : kH;
    packW_k<<<dim3(8, 1), 32, 0, stream>>>(Pm(56 + 2 * l), dinl, kH2 - kDkin, tileP(predB + l * 8));
  }
  packW_k<<<dim3(8, 4), 32, 0, stream>>>(Pm(66), kH2, kH2, tileP(f1B));
  packW_k<<<dim3(8, 4), 32, 0, stream>>>(Pm(68), kH2, kH2, tileP(f2B));
  packW_k<<<dim3(1, 4), 32, 0, stream>>>(Pm(70), kH2, 2,   tileP(f3B));

  const float invN = 1.f / (float)kN;
  const float invG = 1.f / (float)kNG;

  // hidden[0] pooling sums (from x, zero-padded) + kinematics max + layer-0 prep
  graphsum_k<<<kNG * kH / 256, 256, 0, stream>>>(x, kDin, kDin, GS);
  kinmax_k<<<kNG * kDkin / 256, 256, 0, stream>>>(kin, HCAT);
  xprep_k<<<(size_t)kN * kH / 256, 256, 0, stream>>>(x, GS, ZB);

  for (int l = 0; l < 4; ++l) {
    const int b = l * 14;
    const float *B1 = Pm(b + 1), *B2 = Pm(b + 3), *B3 = Pm(b + 5);
    const float *G1 = Pm(b + 6), *Be1 = Pm(b + 7), *G2 = Pm(b + 8), *Be2 = Pm(b + 9);
    const float *GA = Pm(b + 10), *BA = Pm(b + 11), *GO = Pm(b + 12), *BO = Pm(b + 13);
    float* gsum_l = GS + (size_t)l * kN;

    // mlp linear1 -> z1, stats.  Layer 0: A pre-computed (RAW); layers 1-3: agg fused.
    if (l == 0)
      gemm32_k<2><<<kN / 128, 256, 0, stream>>>(ZB, gsum_l, nullptr, nullptr,
                                                tileP(ginB(l, 0)), B1, ZA, PART);
    else
      gemm32_k<0><<<kN / 128, 256, 0, stream>>>(Hbuf, gsum_l, nullptr, nullptr,
                                                tileP(ginB(l, 0)), B1, ZA, PART);
    bn_reduce_k<<<1, 64, 0, stream>>>(PART, kN / 128, kH, invN, G1, Be1, SCA, SHA);
    // linear2 (BN1+ReLU fused) -> z2, stats
    gemm32_k<1><<<kN / 128, 256, 0, stream>>>(ZA, nullptr, SCA, SHA,
                                              tileP(ginB(l, 1)), B2, ZB, PART);
    bn_reduce_k<<<1, 64, 0, stream>>>(PART, kN / 128, kH, invN, G2, Be2, SCA, SHA);
    // linear3 (BN2+ReLU fused) -> z3, stats (for apply_bn)
    gemm32_k<1><<<kN / 128, 256, 0, stream>>>(ZB, nullptr, SCA, SHA,
                                              tileP(ginB(l, 2)), B3, ZA, PART);
    bn_reduce_k<<<1, 64, 0, stream>>>(PART, kN / 128, kH, invN, GA, BA, SCA, SHA);
    // outer BN needs stats of relu(apply_bn(z3)) -- computed without materializing it
    relu_stats_k<<<kN / 256, 256, 0, stream>>>(ZA, SCA, SHA, PART, 256);
    bn_reduce_k<<<1, 64, 0, stream>>>(PART, kN / 256, kH, invN, GO, BO, SCB, SHB);
    // h_next = relu(outer_bn(relu(apply_bn(z3))))
    finalize_k<<<(size_t)kN * kH / 256, 256, 0, stream>>>(ZA, SCA, SHA, SCB, SHB, Hbuf);
    // per-graph sums of h_next (next layer agg + pooling)
    graphsum_k<<<kNG * kH / 256, 256, 0, stream>>>(Hbuf, kH, kH, GS + (size_t)(l + 1) * kN);
  }

  // prediction head: hcat[:,0:120]
  pred_k<<<kNG / 128, 256, 0, stream>>>(GS, tileP(predB),
      Pm(57), Pm(59), Pm(61), Pm(63), Pm(65), HCAT);

  // final MLP over hcat [4096,128]
  const float *FB1 = Pm(67), *FB2 = Pm(69), *FB3 = Pm(71);
  const float *FG1 = Pm(72), *FBe1 = Pm(73), *FG2 = Pm(74), *FBe2 = Pm(75);

  gemm128_k<0, true><<<kNG / 128, 256, 0, stream>>>(HCAT, nullptr, nullptr, tileP(f1B), FB1,
                                                    ZF1, kH2, kH2, kH2, 8, PART);
  bn_reduce_k<<<1, 256, 0, stream>>>(PART, kNG / 128, kH2, invG, FG1, FBe1, SCA, SHA);
  gemm128_k<1, true><<<kNG / 128, 256, 0, stream>>>(ZF1, SCA, SHA, tileP(f2B), FB2,
                                                    ZF2, kH2, kH2, kH2, 8, PART);
  bn_reduce_k<<<1, 256, 0, stream>>>(PART, kNG / 128, kH2, invG, FG2, FBe2, SCB, SHB);
  // last linear 128->2 (BN2+ReLU fused), write straight to d_out [4096,2]
  gemm128_k<1, false><<<kNG / 128, 256, 0, stream>>>(ZF2, SCB, SHB, tileP(f3B), FB3,
                                                     (float*)d_out, 2, 2, 2, 1, nullptr);
}
